// Attention_2765958939263
// MI455X (gfx1250) — matrix operand order repaired and checked by reading, not compiled, not run on hardware
//
#include <hip/hip_runtime.h>
#include <math.h>

// ---------------------------------------------------------------------------
// MI455X (gfx1250) fused windowed channel-attention.
//   k1: qkv GEMM  (bf16 WMMA, 128x128 tile, double-buffered LDS) -> ws (bf16)
//   k2: fused dwconv3x3 + l2norm + channel attention (bf16 WMMA) -> ws (bf16)
//   k3: proj GEMM (bf16 WMMA) + bias                             -> d_out (f32)
// ---------------------------------------------------------------------------

typedef __attribute__((ext_vector_type(16))) __bf16 v16bf;
typedef __attribute__((ext_vector_type(8)))  float   v8f;
typedef __attribute__((ext_vector_type(4)))  float   v4f;
typedef __attribute__((ext_vector_type(8)))  __bf16  bh8;
typedef __attribute__((ext_vector_type(4)))  __bf16  bh4;

static constexpr int kDim   = 384;
static constexpr int kHeads = 8;
static constexpr int kCph   = 48;     // channels per head
static constexpr int kWs    = 8;      // window size
static constexpr int kNwin  = 64;     // pixels per window (8*8)
static constexpr int kB     = 2;
static constexpr int kH     = 256;
static constexpr int kW     = 256;
static constexpr int kQkvC  = 3 * kDim;   // 1152
static constexpr size_t kHW = (size_t)kH * kW;

// ---- helpers ---------------------------------------------------------------

__device__ inline void storeOut(float* p, size_t i, float v)  { p[i] = v; }
__device__ inline void storeOut(__bf16* p, size_t i, float v) { p[i] = (__bf16)v; }

// A fragment (16x32 bf16, row-major [m][k] in LDS, K-contiguous):
//   lanes 0-15 : comps 0..7 -> K 0..7,  comps 8..15 -> K 16..23
//   lanes16-31 : comps 0..7 -> K 8..15, comps 8..15 -> K 24..31
__device__ inline v16bf ldfragA(const __bf16* rowBase, bool hi) {
  bh8 lo = *(const bh8*)(rowBase + (hi ? 8 : 0));
  bh8 h8 = *(const bh8*)(rowBase + (hi ? 24 : 16));
  v16bf r;
#pragma unroll
  for (int i = 0; i < 8; ++i) { r[i] = lo[i]; r[8 + i] = h8[i]; }
  return r;
}
// B fragment (32x16 bf16), stored transposed [n][k] in LDS, K-contiguous:
//   lanes 0-15 : comps 0..7 -> K 0..7,  comps 8..15 -> K 16..23
//   lanes16-31 : comps 0..7 -> K 8..15, comps 8..15 -> K 24..31
__device__ inline v16bf ldfragB(const __bf16* rowBase, bool hi) {
  bh8 lo = *(const bh8*)(rowBase + (hi ? 8 : 0));
  bh8 h8 = *(const bh8*)(rowBase + (hi ? 24 : 16));
  v16bf r;
#pragma unroll
  for (int i = 0; i < 8; ++i) { r[i] = lo[i]; r[8 + i] = h8[i]; }
  return r;
}

// B-tile loaders: global [K,N] slab -> LDS transposed [n][k], bf16.
// f32 activations: 1024 x float4 (b128) loads.
__device__ inline void loadBtile(const float* Bp, __bf16 (*sB)[40], int kb, int n0,
                                 int N, int tid) {
#pragma unroll
  for (int i = 0; i < 4; ++i) {
    int idx = tid + i * 256;                 // 0..1023
    int k = idx >> 5, nq = idx & 31;
    v4f v = *(const v4f*)&Bp[(size_t)(kb + k) * N + n0 + nq * 4];
#pragma unroll
    for (int j = 0; j < 4; ++j) sB[nq * 4 + j][k] = (__bf16)v[j];
  }
}
// bf16 activations: 512 x 8xbf16 (b128) loads.
__device__ inline void loadBtile(const __bf16* Bp, __bf16 (*sB)[40], int kb, int n0,
                                 int N, int tid) {
#pragma unroll
  for (int i = 0; i < 2; ++i) {
    int idx = tid + i * 256;                 // 0..511
    int k = idx >> 4, nq = idx & 15;
    bh8 v = *(const bh8*)&Bp[(size_t)(kb + k) * N + n0 + nq * 8];
#pragma unroll
    for (int j = 0; j < 8; ++j) sB[nq * 8 + j][k] = v[j];
  }
}

// ---- generic bf16 WMMA GEMM: C[m,n] = sum_k A[m,k]*B[k,n] (+bias) ----------
// Block tile 128x128, BK=32, 256 threads = 8 waves in a 4x2 grid, 32x64/wave.
// Double-buffered LDS: one barrier per K-step, global loads overlap WMMA.

template <typename TIn, typename TOut, bool HAS_BIAS>
__global__ __launch_bounds__(256)
void gemm_wmma_bf16(const float* __restrict__ A, const TIn* __restrict__ Bact,
                    TOut* __restrict__ C, const float* __restrict__ bias,
                    int M, int K, int N) {
  constexpr int BM = 128, BN = 128, LK = 40;  // LK: padded K stride in LDS
  const int m0 = blockIdx.x * BM;
  const int n0 = blockIdx.y * BN;
  Bact += (size_t)blockIdx.z * K * N;
  C    += (size_t)blockIdx.z * M * N;

  __shared__ __align__(16) __bf16 sA[2][BM][LK];   // [m][k]
  __shared__ __align__(16) __bf16 sB[2][BN][LK];   // [n][k] (transposed)

  const int tid  = threadIdx.x;
  const int lane = tid & 31, wv = tid >> 5;
  const int lm   = lane & 15;
  const bool hi  = lane >= 16;
  const int wm   = wv >> 1, wn = wv & 1;           // 4 x 2 wave grid

  // A-tile loader: 128x32 f32 -> bf16, 1024 x float4 (b128) loads.
  auto loadA = [&](int kb, int buf) {
#pragma unroll
    for (int i = 0; i < 4; ++i) {
      int idx = tid + i * 256;                     // 0..1023
      int m = idx >> 3, q = idx & 7;
      v4f v = *(const v4f*)&A[(size_t)(m0 + m) * K + kb + q * 4];
      bh4 h;
#pragma unroll
      for (int j = 0; j < 4; ++j) h[j] = (__bf16)v[j];
      *(bh4*)&sA[buf][m][q * 4] = h;
    }
  };

  loadA(0, 0);
  loadBtile(Bact, sB[0], 0, n0, N, tid);
  __syncthreads();

  v8f acc[2][4] = {};
  const int NT = K / 32;
  for (int it = 0; it < NT; ++it) {
    const int cur = it & 1;
    if (it + 1 < NT) {                 // prefetch next K-tile into other buffer
      loadA((it + 1) * 32, cur ^ 1);
      loadBtile(Bact, sB[cur ^ 1], (it + 1) * 32, n0, N, tid);
    }
    v16bf af[2], bfr[4];
#pragma unroll
    for (int mi = 0; mi < 2; ++mi)
      af[mi] = ldfragA(&sA[cur][wm * 32 + mi * 16 + lm][0], hi);
#pragma unroll
    for (int ni = 0; ni < 4; ++ni)
      bfr[ni] = ldfragB(&sB[cur][wn * 64 + ni * 16 + lm][0], hi);
#pragma unroll
    for (int mi = 0; mi < 2; ++mi)
#pragma unroll
      for (int ni = 0; ni < 4; ++ni)
        acc[mi][ni] = __builtin_amdgcn_wmma_f32_16x16x32_bf16(
            false, af[mi], false, bfr[ni], (short)0, acc[mi][ni], false, false);
    __syncthreads();
  }

#pragma unroll
  for (int mi = 0; mi < 2; ++mi)
#pragma unroll
    for (int ni = 0; ni < 4; ++ni)
#pragma unroll
      for (int j = 0; j < 8; ++j) {
        int m = m0 + wm * 32 + mi * 16 + j + (hi ? 8 : 0);
        int n = n0 + wn * 64 + ni * 16 + lm;
        float v = acc[mi][ni][j];
        if (HAS_BIAS) v += bias[m];
        storeOut(C, (size_t)m * N + n, v);
      }
}

// ---- fused depthwise-3x3 + l2norm + channel attention ----------------------
// One block (128 threads, 4 waves) per (b, window, head).

__global__ __launch_bounds__(128)
void attn_fused(const __bf16* __restrict__ qkv, const float* __restrict__ dww,
                const float* __restrict__ temperature, __bf16* __restrict__ outb) {
  __shared__ __align__(16) __bf16 sQ[kCph][kNwin];     // q (l2-normed), [c][n]
  __shared__ __align__(16) __bf16 sK[kCph][kNwin];     // k (l2-normed), [d][n]
  __shared__ __align__(16) __bf16 sVT[kNwin][kNwin];   // v transposed [n][d], d>=48 zero
  __shared__ __align__(16) __bf16 sP[kCph][kNwin];     // softmax probs [c][d], d>=48 zero
  __shared__ __align__(16) float  sL[kCph][kNwin];     // f32 logits
  __shared__ __align__(16) __bf16 sHalo[3 * kCph][100]; // 10x10 halo per channel

  const int tid  = threadIdx.x;
  const int head = blockIdx.y;
  const int b    = blockIdx.z;
  const int h1   = blockIdx.x >> 5;
  const int w1   = blockIdx.x & 31;
  const int y0   = h1 * kWs, x0 = w1 * kWs;

  // Stage A: 144-channel x 10x10 bf16 halo -> LDS (zero-padded at borders)
  for (int idx = tid; idx < 3 * kCph * 100; idx += 128) {
    int ch = idx / 100, p = idx - ch * 100;
    int iy = p / 10, ix = p - iy * 10;
    int z = ch / kCph, c = ch - z * kCph;
    int gch = z * kDim + head * kCph + c;
    int gy = y0 - 1 + iy, gx = x0 - 1 + ix;
    __bf16 v = (__bf16)0.0f;
    if ((unsigned)gy < (unsigned)kH && (unsigned)gx < (unsigned)kW)
      v = qkv[((size_t)b * kQkvC + gch) * kHW + (size_t)gy * kW + gx];
    sHalo[ch][p] = v;
  }
  __syncthreads();

  // Stage B: depthwise 3x3 (cross-correlation, pad=1); l2-normalize q,k rows;
  //          transpose v into sVT.
  for (int r = tid; r < 3 * kCph; r += 128) {
    int z = r / kCph, c = r - z * kCph;
    int gch = z * kDim + head * kCph + c;
    float w[9];
#pragma unroll
    for (int i = 0; i < 9; ++i) w[i] = dww[gch * 9 + i];
    float ss = 0.f;
    for (int n = 0; n < kNwin; ++n) {
      int hh = n >> 3, ww = n & 7;
      float a = 0.f;
#pragma unroll
      for (int ky = 0; ky < 3; ++ky)
#pragma unroll
        for (int kx = 0; kx < 3; ++kx)
          a += w[ky * 3 + kx] * (float)sHalo[r][(hh + ky) * 10 + (ww + kx)];
      ss += a * a;
      if (z == 0)       sQ[c][n] = (__bf16)a;
      else if (z == 1)  sK[c][n] = (__bf16)a;
      else              sVT[n][c] = (__bf16)a;
    }
    if (z < 2) {
      float inv = 1.f / fmaxf(sqrtf(ss), 1e-12f);
      __bf16* row = (z == 0) ? &sQ[c][0] : &sK[c][0];
      for (int n = 0; n < kNwin; ++n) row[n] = (__bf16)((float)row[n] * inv);
    }
  }
  // zero-pad sVT columns d = 48..63 (K padding for second WMMA)
  for (int idx = tid; idx < kNwin * 16; idx += 128)
    sVT[idx >> 4][kCph + (idx & 15)] = (__bf16)0.0f;
  __syncthreads();

  const int lane = tid & 31, wv = tid >> 5;
  const int lm   = lane & 15;
  const bool hi  = lane >= 16;
  const float tscale = temperature[head];

  // Stage C: logits = (q @ k^T) * temp  -> 48x48, 3x3 tiles over 4 waves
  for (int t = wv; t < 9; t += 4) {
    int mt = t / 3, nt = t - mt * 3;
    v8f acc = {};
#pragma unroll
    for (int kc = 0; kc < 2; ++kc) {
      v16bf a  = ldfragA(&sQ[mt * 16 + lm][kc * 32], hi);
      v16bf bm = ldfragB(&sK[nt * 16 + lm][kc * 32], hi);
      acc = __builtin_amdgcn_wmma_f32_16x16x32_bf16(false, a, false, bm, (short)0, acc,
                                                    false, false);
    }
#pragma unroll
    for (int j = 0; j < 8; ++j) {
      int m = mt * 16 + j + (hi ? 8 : 0);   // < 48 by construction
      int n = nt * 16 + lm;                 // < 48 by construction
      sL[m][n] = acc[j] * tscale;
    }
  }
  __syncthreads();

  // Stage D: row softmax over d=0..47, store bf16 probs, pad d=48..63 with 0
  if (tid < kCph) {
    float mx = -3.4e38f;
    for (int d = 0; d < kCph; ++d) mx = fmaxf(mx, sL[tid][d]);
    float s = 0.f;
    for (int d = 0; d < kCph; ++d) s += __expf(sL[tid][d] - mx);
    float rs = 1.f / s;
    for (int d = 0; d < kCph; ++d) sP[tid][d] = (__bf16)(__expf(sL[tid][d] - mx) * rs);
    for (int d = kCph; d < kNwin; ++d) sP[tid][d] = (__bf16)0.0f;
  }
  __syncthreads();

  // Stage E: out = P @ V  (48x64, K=64 incl. zero pad), 3x4 tiles over 4 waves
  for (int t = wv; t < 12; t += 4) {
    int mt = t >> 2, nt = t & 3;
    v8f acc = {};
#pragma unroll
    for (int kc = 0; kc < 2; ++kc) {
      v16bf a  = ldfragA(&sP[mt * 16 + lm][kc * 32], hi);
      v16bf bm = ldfragB(&sVT[nt * 16 + lm][kc * 32], hi);
      acc = __builtin_amdgcn_wmma_f32_16x16x32_bf16(false, a, false, bm, (short)0, acc,
                                                    false, false);
    }
#pragma unroll
    for (int j = 0; j < 8; ++j) {
      int m = mt * 16 + j + (hi ? 8 : 0);   // channel within head, 0..47
      int n = nt * 16 + lm;                 // window pixel, 0..63
      int hh = n >> 3, ww = n & 7;
      outb[((size_t)b * kDim + head * kCph + m) * kHW + (size_t)(y0 + hh) * kW +
           (x0 + ww)] = (__bf16)acc[j];
    }
  }
}

// ---------------------------------------------------------------------------

extern "C" void kernel_launch(void* const* d_in, const int* in_sizes, int n_in,
                              void* d_out, int out_size, void* d_ws, size_t ws_size,
                              hipStream_t stream) {
  const float* x      = (const float*)d_in[0];   // [2,384,256,256]
  const float* qkv_w  = (const float*)d_in[1];   // [1152,384]
  const float* dw_w   = (const float*)d_in[2];   // [1152,1,3,3]
  const float* temp   = (const float*)d_in[3];   // [8,1,1]
  const float* proj_w = (const float*)d_in[4];   // [384,384]
  const float* proj_b = (const float*)d_in[5];   // [384]
  float* out = (float*)d_out;

  // workspace: qkv (bf16, 302 MB) + attention output (bf16, 101 MB)
  __bf16* qkvbuf  = (__bf16*)d_ws;
  __bf16* attnbuf = qkvbuf + (size_t)kB * kQkvC * kHW;

  // 1) qkv 1x1 conv: [1152x384] x [384x65536] per batch
  dim3 g1(kQkvC / 128, (int)(kHW / 128), kB);
  gemm_wmma_bf16<float, __bf16, false><<<g1, 256, 0, stream>>>(
      qkv_w, x, qkvbuf, nullptr, kQkvC, kDim, (int)kHW);

  // 2) fused dwconv + windowed channel attention: one block per (win, head, b)
  dim3 g2((kH / kWs) * (kW / kWs), kHeads, kB);
  attn_fused<<<g2, 128, 0, stream>>>(qkvbuf, dw_w, temp, attnbuf);

  // 3) proj 1x1 conv + bias: [384x384] x [384x65536] per batch -> f32 out
  dim3 g3(kDim / 128, (int)(kHW / 128), kB);
  gemm_wmma_bf16<__bf16, float, true><<<g3, 256, 0, stream>>>(
      proj_w, attnbuf, out, proj_b, kDim, kDim, (int)kHW);
}
